// HGCODE_20933670601184
// MI455X (gfx1250) — matrix operand order repaired and checked by reading, not compiled, not run on hardware
//
#include <hip/hip_runtime.h>
#include <hip/hip_bf16.h>

// ---------------------------------------------------------------------------
// CDNA5 (gfx1250) implementation of the hierarchical GNN-ODE-GRU scan.
// All dense matmuls use v_wmma_f32_16x16x32_f16 (wave32 WMMA, f32 accum).
// GEMM: weights (B matrix) staged once per workgroup into LDS; each wave owns
// a 16-row strip across the full N, issuing NT back-to-back WMMAs per k-step.
// ---------------------------------------------------------------------------

typedef _Float16 half_t;
typedef __attribute__((ext_vector_type(16))) _Float16 v16h;
typedef __attribute__((ext_vector_type(8)))  _Float16 v8h;
typedef __attribute__((ext_vector_type(8)))  float    v8f;

#define Bc   16
#define Tc   128
#define Pc   8
#define Jc   26
#define Dc   64
#define Hc   128
#define NGc  4
#define DINc 2

// ---------------------------------------------------------------------------
// Input index map (setup_inputs dict order, params flattened as a JAX pytree:
// dict keys alphabetical, lists/tuples in order).
// ---------------------------------------------------------------------------
enum {
  IN_X2D = 0, IN_MASK, IN_G1, IN_G2, IN_G3, IN_ADJ1, IN_ADJ2, IN_ADJ3,
  IN_WOUT, IN_BOUT,
  G1C0_WN, G1C0_WR, G1C0_WZ, G1C0_BN, G1C0_BR, G1C0_BZ,
  G1C1_WN, G1C1_WR, G1C1_WZ, G1C1_BN, G1C1_BR, G1C1_BZ,
  G2C0_WN, G2C0_WR, G2C0_WZ, G2C0_BN, G2C0_BR, G2C0_BZ,
  G2C1_WN, G2C1_WR, G2C1_WZ, G2C1_BN, G2C1_BR, G2C1_BZ,
  G3C0_WN, G3C0_WR, G3C0_WZ, G3C0_BN, G3C0_BR, G3C0_BZ,
  G3C1_WN, G3C1_WR, G3C1_WZ, G3C1_BN, G3C1_BR, G3C1_BZ,
  IN_H0,
  O1_WO, O1_BO, O1_W0, O1_B0, O1_W1, O1_B1,
  O2_WO, O2_BO, O2_W0, O2_B0, O2_W1, O2_B1,
  O3_WO, O3_BO, O3_W0, O3_B0, O3_W1, O3_B1
};

// Generic strided-source descriptor:
//   addr(blk, node, k) = p + (blk/div)*hi + (blk%div)*lo + node*node_s + k
struct Src { const float* p; int w; int div; int hi; int lo; int node; };

// ---------------------------------------------------------------------------
// Weight conversion: fp32 W[K x N] (row-major) -> fragment-swizzled fp16.
// B-fragment layout for v_wmma_f32_16x16x32_f16 (ISA 7.12.2): lane l holds
// column n = (l&15); k_local = (e<8 ? e : e+8) + (l>=16 ? 8 : 0) for element e of v16h.
// Stored so each lane's 16 halves are contiguous (32B => 2x b128 loads).
// ---------------------------------------------------------------------------
__global__ void k_conv_w(const float* __restrict__ W, half_t* __restrict__ dst,
                         int K, int N, int Kpad) {
  int idx = blockIdx.x * blockDim.x + threadIdx.x;
  if (idx >= Kpad * N) return;
  int e    = idx & 15;
  int lane = (idx >> 4) & 31;
  int tile = idx >> 9;
  int NT = N >> 4;
  int nt = tile % NT;
  int kt = tile / NT;
  int n  = nt * 16 + (lane & 15);
  int k  = kt * 32 + ((lane >> 4) << 3) + (e & 7) + ((e >> 3) << 4);
  float v = (k < K) ? W[(size_t)k * N + n] : 0.0f;
  dst[idx] = (half_t)v;
}

// ---------------------------------------------------------------------------
// Fused gather/broadcast/concat/gate/adjacency-aggregate -> fp16 A matrix.
//   inp[blk, v, :] = concat(s0, s1, s2*gate?)   (widths w0, w1, w2)
//   A[blk*NN + u, k] = sum_v adj[(blk%adjcnt)][u][v] * inp[blk, v, k]
// Pad columns (k >= w0+w1+w2) are written as zero.
// ---------------------------------------------------------------------------
__device__ __forceinline__ float src_read(const Src& s, int blk, int v, int k) {
  return s.p[(size_t)(blk / s.div) * s.hi + (size_t)(blk % s.div) * s.lo +
             (size_t)v * s.node + k];
}

__global__ void k_prep_A(half_t* __restrict__ A, int Kpad, int NB, int NN,
                         const float* __restrict__ adj, int adjcnt,
                         Src s0, Src s1, Src s2, Src g, int use_gate) {
  int idx = blockIdx.x * blockDim.x + threadIdx.x;
  int total = NB * NN * Kpad;
  if (idx >= total) return;
  int k   = idx % Kpad;
  int ru  = idx / Kpad;
  int u   = ru % NN;
  int blk = ru / NN;
  int w01  = s0.w + s1.w;
  int wtot = w01 + s2.w;
  float out = 0.0f;
  if (k < wtot) {
    const float* arow = adj + (size_t)(blk % adjcnt) * NN * NN + (size_t)u * NN;
    for (int v = 0; v < NN; ++v) {
      float x;
      if (k < s0.w)      x = src_read(s0, blk, v, k);
      else if (k < w01)  x = src_read(s1, blk, v, k - s0.w);
      else {
        x = src_read(s2, blk, v, k - w01);
        if (use_gate) x *= src_read(g, blk, v, k - w01);
      }
      out += arow[v] * x;
    }
  }
  A[(size_t)ru * Kpad + k] = (half_t)out;
}

// ---------------------------------------------------------------------------
// WMMA GEMM: C[M x N] = epilogue(A[M x Kpad] @ W + bias)
//   mode 0: store   1: tanh   2: sigmoid   3: C = addsrc + scale*raw (axpy)
// Block = 128 threads = 4 waves. The full swizzled weight matrix (<= 32 KB)
// is cooperatively staged into LDS once; each wave then owns one 16-row
// M-tile and sweeps all NT column tiles (NT is a template constant so the
// v8f accumulators stay in VGPRs). Per k-step: 1 A fragment (2x b128 from
// global, L2-resident) then NT back-to-back v_wmma fed by ds_load_b128.
//
// A fragment (ISA 7.12.2): lane l holds row m=(l&15);
//   element e: k_local = (e<8 ? e : e+8) + (l>=16 ? 8 : 0)
// C/D fragment: lane l: n=(l&15); element r: m = r + (l>=16 ? 8 : 0).
// Max LDS: KT*NT*512 halves = (128/32)*8*512 = 16384 halves = 32 KB.
// ---------------------------------------------------------------------------
template <int NT>
__global__ void k_wmma_gemm(const half_t* __restrict__ A, int Kpad,
                            const half_t* __restrict__ Wsw,
                            const float* __restrict__ bias,
                            int N, int MT,
                            float* __restrict__ C,
                            const float* __restrict__ addsrc,
                            float scale, int mode) {
  __shared__ half_t Bs[16384];
  int tid = (int)threadIdx.x;
  int KT = Kpad >> 5;

  // Cooperative stage of the swizzled weights (already fragment-contiguous).
  int chunks = (KT * NT * 512) >> 3;  // 8-half (16B) chunks
  for (int i = tid; i < chunks; i += (int)blockDim.x) {
    *reinterpret_cast<v8h*>(Bs + (size_t)i * 8) =
        *reinterpret_cast<const v8h*>(Wsw + (size_t)i * 8);
  }
  __syncthreads();

  int wave = tid >> 5;
  int mt = (int)blockIdx.x * 4 + wave;
  if (mt >= MT) return;

  int lane = tid & 31;
  int lo16 = lane & 15;
  int hi   = lane >> 4;

  v8f acc[NT];
#pragma unroll
  for (int nt = 0; nt < NT; ++nt) acc[nt] = (v8f){};

  const half_t* arow = A + (size_t)(mt * 16 + lo16) * Kpad + hi * 8;
  for (int kt = 0; kt < KT; ++kt) {
    union { v16h v; v8h h[2]; } a;
    a.h[0] = *reinterpret_cast<const v8h*>(arow + kt * 32);
    a.h[1] = *reinterpret_cast<const v8h*>(arow + kt * 32 + 16);
    const half_t* brow = Bs + (size_t)kt * NT * 512 + lane * 16;
#pragma unroll
    for (int nt = 0; nt < NT; ++nt) {
      union { v16h v; v8h h[2]; } b;
      b.h[0] = *reinterpret_cast<const v8h*>(brow + nt * 512);
      b.h[1] = *reinterpret_cast<const v8h*>(brow + nt * 512 + 8);
      acc[nt] = __builtin_amdgcn_wmma_f32_16x16x32_f16(false, a.v, false, b.v,
                                                       (short)0, acc[nt],
                                                       false, false);
    }
  }

  int row0 = mt * 16 + hi * 8;
#pragma unroll
  for (int nt = 0; nt < NT; ++nt) {
    int n = nt * 16 + lo16;
    float bv = bias[n];
#pragma unroll
    for (int r = 0; r < 8; ++r) {
      float v = acc[nt][r] + bv;
      if (mode == 1)      v = tanhf(v);
      else if (mode == 2) v = 1.0f / (1.0f + __expf(-v));
      size_t o = (size_t)(row0 + r) * N + n;
      if (mode == 3) C[o] = addsrc[o] + scale * v;
      else           C[o] = v;
    }
  }
}

// ---------------------------------------------------------------------------
// Elementwise helpers.
// ---------------------------------------------------------------------------
__global__ void k_init_h(float* __restrict__ H, const float* __restrict__ h0,
                         int total) {
  int i = blockIdx.x * blockDim.x + threadIdx.x;
  if (i < total) H[i] = h0[i & (Dc - 1)];
}

// xs[b,p,j,c] = x2d[b,t,p,j,c] * mask[b,t,p,j,0]
__global__ void k_make_xs(const float* __restrict__ x2d,
                          const float* __restrict__ mask,
                          float* __restrict__ xs, int t) {
  int idx = blockIdx.x * blockDim.x + threadIdx.x;
  if (idx >= Bc * Pc * Jc * DINc) return;
  int c  = idx & 1;
  int j  = (idx >> 1) % Jc;
  int bp = (idx >> 1) / Jc;
  int b = bp / Pc, p = bp % Pc;
  size_t base = (((size_t)b * Tc + t) * Pc + p) * Jc + j;
  xs[idx] = x2d[base * DINc + c] * mask[base];
}

// Split H(b,p,26,64) into group buffers h11(b,p,64), h12(b,p,5,64), h13(b,p,20,64).
__global__ void k_gather_h(const float* __restrict__ H, float* __restrict__ h11,
                           float* __restrict__ h12, float* __restrict__ h13) {
  int idx = blockIdx.x * blockDim.x + threadIdx.x;
  if (idx >= Bc * Pc * Jc * Dc) return;
  int d  = idx & 63;
  int j  = (idx >> 6) % Jc;
  int bp = (idx >> 6) / Jc;
  float v = H[idx];
  if (j == 0)      h11[(size_t)bp * Dc + d] = v;
  else if (j < 6)  h12[((size_t)bp * 5  + (j - 1)) * Dc + d] = v;
  else             h13[((size_t)bp * 20 + (j - 6)) * Dc + d] = v;
}

__global__ void k_scatter_h(float* __restrict__ H, const float* __restrict__ h21,
                            const float* __restrict__ h22,
                            const float* __restrict__ h23) {
  int idx = blockIdx.x * blockDim.x + threadIdx.x;
  if (idx >= Bc * Pc * Jc * Dc) return;
  int d  = idx & 63;
  int j  = (idx >> 6) % Jc;
  int bp = (idx >> 6) / Jc;
  float v;
  if (j == 0)      v = h21[(size_t)bp * Dc + d];
  else if (j < 6)  v = h22[((size_t)bp * 5  + (j - 1)) * Dc + d];
  else             v = h23[((size_t)bp * 20 + (j - 6)) * Dc + d];
  H[idx] = v;
}

__global__ void k_gru_combine(float* __restrict__ out, const float* __restrict__ z,
                              const float* __restrict__ h, const float* __restrict__ n,
                              int total) {
  int i = blockIdx.x * blockDim.x + threadIdx.x;
  if (i < total) {
    float zi = z[i];
    out[i] = zi * h[i] + (1.0f - zi) * n[i];
  }
}

// out[b,t,p,j,:] = H[b,p,j,:] @ Wout(64x3) + bout
__global__ void k_proj_out(const float* __restrict__ H, const float* __restrict__ Wout,
                           const float* __restrict__ bout, float* __restrict__ out,
                           int t) {
  int idx = blockIdx.x * blockDim.x + threadIdx.x;
  if (idx >= Bc * Pc * Jc * 3) return;
  int o  = idx % 3;
  int j  = (idx / 3) % Jc;
  int bp = (idx / 3) / Jc;
  int b = bp / Pc, p = bp % Pc;
  const float* h = H + ((size_t)bp * Jc + j) * Dc;
  float acc = bout[o];
  for (int d = 0; d < Dc; ++d) acc += h[d] * Wout[d * 3 + o];
  out[((((size_t)b * Tc + t) * Pc + p) * Jc + j) * 3 + o] = acc;
}

// ---------------------------------------------------------------------------
// Host-side orchestration.
// ---------------------------------------------------------------------------
static inline int cdiv(int a, int b) { return (a + b - 1) / b; }

static inline void* bump(char*& p, size_t bytes) {
  void* r = (void*)p;
  p += (bytes + 255) & ~(size_t)255;
  return r;
}

struct Ctx {
  hipStream_t s;
  half_t* A16;
  float *z1, *z2, *rb, *zb, *nb;
};

static void launch_prep(const Ctx& c, int Kpad, int NB, int NN,
                        const float* adj, int adjcnt,
                        Src s0, Src s1, Src s2, Src g, int use_gate) {
  int tot = NB * NN * Kpad;
  k_prep_A<<<cdiv(tot, 256), 256, 0, c.s>>>(c.A16, Kpad, NB, NN, adj, adjcnt,
                                            s0, s1, s2, g, use_gate);
}

static void launch_gemm(const Ctx& c, int M, int Kpad, int N, const half_t* Wsw,
                        const float* bias, float* Cout, const float* addsrc,
                        float scale, int mode) {
  int MT = M / 16;
  int blocks = cdiv(MT, 4);
  if (N == 128) {
    k_wmma_gemm<8><<<blocks, 128, 0, c.s>>>(c.A16, Kpad, Wsw, bias, N, MT,
                                            Cout, addsrc, scale, mode);
  } else {
    k_wmma_gemm<4><<<blocks, 128, 0, c.s>>>(c.A16, Kpad, Wsw, bias, N, MT,
                                            Cout, addsrc, scale, mode);
  }
}

static const Src ZS = {nullptr, 0, 1, 0, 0, 0};

// GNN ODE block: h_dst += scale * (agg->W0,tanh -> agg->W1,tanh -> agg->Wo)
static void gnn_block(const Ctx& c, Src i0, Src i1, const float* adj, int adjcnt,
                      int NB, int NN, int Kin,
                      const half_t* W0, const float* b0,
                      const half_t* W1, const float* b1,
                      const half_t* Wo, const float* bo,
                      float* hdst, float scale) {
  int M = NB * NN;
  launch_prep(c, Kin, NB, NN, adj, adjcnt, i0, i1, ZS, ZS, 0);
  launch_gemm(c, M, Kin, Hc, W0, b0, c.z1, nullptr, 0.f, 1);
  Src sz1 = {c.z1, Hc, 1, NN * Hc, 0, Hc};
  launch_prep(c, Hc, NB, NN, adj, adjcnt, sz1, ZS, ZS, ZS, 0);
  launch_gemm(c, M, Hc, Hc, W1, b1, c.z2, nullptr, 0.f, 1);
  Src sz2 = {c.z2, Hc, 1, NN * Hc, 0, Hc};
  launch_prep(c, Hc, NB, NN, adj, adjcnt, sz2, ZS, ZS, ZS, 0);
  launch_gemm(c, M, Hc, Dc, Wo, bo, hdst, hdst, scale, 3);
}

// One GRU cell: out = z*h + (1-z)*n with r,z,n from WMMA GEMMs.
static void gru_cell(const Ctx& c, Src s0, Src s1, Src s2, const float* adj,
                     int adjcnt, int NB, int NN, int Kpad,
                     const half_t* Wr, const float* br,
                     const half_t* Wz, const float* bz,
                     const half_t* Wn, const float* bn,
                     const float* hbuf, float* outbuf) {
  int M = NB * NN;
  launch_prep(c, Kpad, NB, NN, adj, adjcnt, s0, s1, s2, ZS, 0);
  launch_gemm(c, M, Kpad, Dc, Wr, br, c.rb, nullptr, 0.f, 2);
  launch_gemm(c, M, Kpad, Dc, Wz, bz, c.zb, nullptr, 0.f, 2);
  Src g = {c.rb, Dc, 1, NN * Dc, 0, Dc};
  launch_prep(c, Kpad, NB, NN, adj, adjcnt, s0, s1, s2, g, 1);
  launch_gemm(c, M, Kpad, Dc, Wn, bn, c.nb, nullptr, 0.f, 1);
  int tot = M * Dc;
  k_gru_combine<<<cdiv(tot, 256), 256, 0, c.s>>>(outbuf, c.zb, hbuf, c.nb, tot);
}

static half_t* conv_weight(hipStream_t s, char*& wp, const void* src,
                           int K, int N, int Kpad) {
  half_t* dst = (half_t*)bump(wp, (size_t)Kpad * N * sizeof(half_t));
  int tot = Kpad * N;
  k_conv_w<<<cdiv(tot, 256), 256, 0, s>>>((const float*)src, dst, K, N, Kpad);
  return dst;
}

extern "C" void kernel_launch(void* const* d_in, const int* in_sizes, int n_in,
                              void* d_out, int out_size, void* d_ws, size_t ws_size,
                              hipStream_t stream) {
  (void)in_sizes; (void)n_in; (void)out_size; (void)ws_size;

  const float* x2d  = (const float*)d_in[IN_X2D];
  const float* mask = (const float*)d_in[IN_MASK];
  const float* adj1 = (const float*)d_in[IN_ADJ1];
  const float* adj2 = (const float*)d_in[IN_ADJ2];
  const float* adj3 = (const float*)d_in[IN_ADJ3];
  const float* Wout = (const float*)d_in[IN_WOUT];
  const float* bout = (const float*)d_in[IN_BOUT];
  const float* h0   = (const float*)d_in[IN_H0];
  float* out = (float*)d_out;

  // ------------------ workspace carve-up (L2-resident, ~10 MB) -------------
  char* wp = (char*)d_ws;
  float* H   = (float*)bump(wp, (size_t)Bc * Pc * Jc * Dc * 4);
  float* h11 = (float*)bump(wp, (size_t)Bc * Pc * Dc * 4);
  float* h12 = (float*)bump(wp, (size_t)Bc * Pc * 5 * Dc * 4);
  float* h13 = (float*)bump(wp, (size_t)Bc * Pc * 20 * Dc * 4);
  float* h21 = (float*)bump(wp, (size_t)Bc * Pc * Dc * 4);
  float* h22 = (float*)bump(wp, (size_t)Bc * Pc * 5 * Dc * 4);
  float* h23 = (float*)bump(wp, (size_t)Bc * Pc * 20 * Dc * 4);
  float* xs  = (float*)bump(wp, (size_t)Bc * Pc * Jc * DINc * 4);
  float* z1  = (float*)bump(wp, (size_t)2560 * Hc * 4);
  float* z2  = (float*)bump(wp, (size_t)2560 * Hc * 4);
  float* rb  = (float*)bump(wp, (size_t)2560 * Dc * 4);
  float* zb  = (float*)bump(wp, (size_t)2560 * Dc * 4);
  float* nb  = (float*)bump(wp, (size_t)2560 * Dc * 4);
  float* cur = (float*)bump(wp, (size_t)2560 * Dc * 4);
  half_t* A16 = (half_t*)bump(wp, (size_t)2560 * 160 * sizeof(half_t));

  Ctx c = {stream, A16, z1, z2, rb, zb, nb};

  // ------------------ one-shot fp16 weight swizzle (idempotent) ------------
  half_t* o1W0 = conv_weight(stream, wp, d_in[O1_W0],  64, 128,  64);
  half_t* o1W1 = conv_weight(stream, wp, d_in[O1_W1], 128, 128, 128);
  half_t* o1Wo = conv_weight(stream, wp, d_in[O1_WO], 128,  64, 128);
  half_t* o2W0 = conv_weight(stream, wp, d_in[O2_W0], 128, 128, 128);
  half_t* o2W1 = conv_weight(stream, wp, d_in[O2_W1], 128, 128, 128);
  half_t* o2Wo = conv_weight(stream, wp, d_in[O2_WO], 128,  64, 128);
  half_t* o3W0 = conv_weight(stream, wp, d_in[O3_W0], 128, 128, 128);
  half_t* o3W1 = conv_weight(stream, wp, d_in[O3_W1], 128, 128, 128);
  half_t* o3Wo = conv_weight(stream, wp, d_in[O3_WO], 128,  64, 128);
  half_t* g1c0Wr = conv_weight(stream, wp, d_in[G1C0_WR],  66, 64,  96);
  half_t* g1c0Wz = conv_weight(stream, wp, d_in[G1C0_WZ],  66, 64,  96);
  half_t* g1c0Wn = conv_weight(stream, wp, d_in[G1C0_WN],  66, 64,  96);
  half_t* g1c1Wr = conv_weight(stream, wp, d_in[G1C1_WR], 128, 64, 128);
  half_t* g1c1Wz = conv_weight(stream, wp, d_in[G1C1_WZ], 128, 64, 128);
  half_t* g1c1Wn = conv_weight(stream, wp, d_in[G1C1_WN], 128, 64, 128);
  half_t* g2c0Wr = conv_weight(stream, wp, d_in[G2C0_WR], 130, 64, 160);
  half_t* g2c0Wz = conv_weight(stream, wp, d_in[G2C0_WZ], 130, 64, 160);
  half_t* g2c0Wn = conv_weight(stream, wp, d_in[G2C0_WN], 130, 64, 160);
  half_t* g2c1Wr = conv_weight(stream, wp, d_in[G2C1_WR], 128, 64, 128);
  half_t* g2c1Wz = conv_weight(stream, wp, d_in[G2C1_WZ], 128, 64, 128);
  half_t* g2c1Wn = conv_weight(stream, wp, d_in[G2C1_WN], 128, 64, 128);
  half_t* g3c0Wr = conv_weight(stream, wp, d_in[G3C0_WR], 130, 64, 160);
  half_t* g3c0Wz = conv_weight(stream, wp, d_in[G3C0_WZ], 130, 64, 160);
  half_t* g3c0Wn = conv_weight(stream, wp, d_in[G3C0_WN], 130, 64, 160);
  half_t* g3c1Wr = conv_weight(stream, wp, d_in[G3C1_WR], 128, 64, 128);
  half_t* g3c1Wz = conv_weight(stream, wp, d_in[G3C1_WZ], 128, 64, 128);
  half_t* g3c1Wn = conv_weight(stream, wp, d_in[G3C1_WN], 128, 64, 128);

  // ------------------ init state -------------------------------------------
  {
    int tot = Bc * Pc * Jc * Dc;
    k_init_h<<<cdiv(tot, 256), 256, 0, stream>>>(H, h0, tot);
  }

  // ------------------ sequential scan over T -------------------------------
  for (int t = 0; t < Tc; ++t) {
    { int tot = Bc * Pc * Jc * DINc;
      k_make_xs<<<cdiv(tot, 256), 256, 0, stream>>>(x2d, mask, xs, t); }
    { int tot = Bc * Pc * Jc * Dc;
      k_gather_h<<<cdiv(tot, 256), 256, 0, stream>>>(H, h11, h12, h13); }

    // ODE level 1: nodes = the P dim, blocks = batch.  h11 += 0.2 * gnn(...)
    Src s_h11 = {h11, Dc, 1, Pc * Dc, 0, Dc};
    gnn_block(c, s_h11, ZS, adj1, 1, Bc, Pc, Dc,
              o1W0, (const float*)d_in[O1_B0], o1W1, (const float*)d_in[O1_B1],
              o1Wo, (const float*)d_in[O1_BO], h11, 2.0f * 2 * 0.05f);

    for (int k1 = 0; k1 < 2; ++k1) {
      // ODE level 2: concat(bcast h11, h12), blocks=(b,p), nodes=5
      Src s_h11b = {h11, Dc, 1, Dc, 0, 0};
      Src s_h12  = {h12, Dc, 1, 5 * Dc, 0, Dc};
      gnn_block(c, s_h11b, s_h12, adj2, 1, Bc * Pc, 5, Hc,
                o2W0, (const float*)d_in[O2_B0], o2W1, (const float*)d_in[O2_B1],
                o2Wo, (const float*)d_in[O2_BO], h12, 2 * 0.05f);
      for (int k2 = 0; k2 < 2; ++k2) {
        // ODE level 3: concat(par = h12 node 1+g bcast, h13), blocks=(b,p,g)
        Src s_par = {h12 + Dc, Dc, 4, 5 * Dc, Dc, 0};
        Src s_h13 = {h13, Dc, 1, 5 * Dc, 0, Dc};
        gnn_block(c, s_par, s_h13, adj3, NGc, Bc * Pc * NGc, 5, Hc,
                  o3W0, (const float*)d_in[O3_B0], o3W1, (const float*)d_in[O3_B1],
                  o3Wo, (const float*)d_in[O3_BO], h13, 0.05f);
      }
    }

    // GRU stack 1 (blocks=b, nodes=P, adj1); h = h11 for both cells.
    Src s_x1   = {xs, DINc, 1, Pc * Jc * DINc, 0, Jc * DINc};
    Src s_h11g = {h11, Dc, 1, Pc * Dc, 0, Dc};
    gru_cell(c, s_x1, ZS, s_h11g, adj1, 1, Bc, Pc, 96,
             g1c0Wr, (const float*)d_in[G1C0_BR], g1c0Wz, (const float*)d_in[G1C0_BZ],
             g1c0Wn, (const float*)d_in[G1C0_BN], h11, cur);
    Src s_cur1 = {cur, Dc, 1, Pc * Dc, 0, Dc};
    gru_cell(c, s_cur1, ZS, s_h11g, adj1, 1, Bc, Pc, 128,
             g1c1Wr, (const float*)d_in[G1C1_BR], g1c1Wz, (const float*)d_in[G1C1_BZ],
             g1c1Wn, (const float*)d_in[G1C1_BN], h11, h21);

    // GRU stack 2 (blocks=(b,p), nodes=5, adj2); h = h12.
    Src s_h21b = {h21, Dc, 1, Dc, 0, 0};
    Src s_x2   = {xs + DINc, DINc, 1, Jc * DINc, 0, DINc};
    Src s_h12g = {h12, Dc, 1, 5 * Dc, 0, Dc};
    gru_cell(c, s_h21b, s_x2, s_h12g, adj2, 1, Bc * Pc, 5, 160,
             g2c0Wr, (const float*)d_in[G2C0_BR], g2c0Wz, (const float*)d_in[G2C0_BZ],
             g2c0Wn, (const float*)d_in[G2C0_BN], h12, cur);
    Src s_cur2 = {cur, Dc, 1, 5 * Dc, 0, Dc};
    gru_cell(c, s_cur2, ZS, s_h12g, adj2, 1, Bc * Pc, 5, 128,
             g2c1Wr, (const float*)d_in[G2C1_BR], g2c1Wz, (const float*)d_in[G2C1_BZ],
             g2c1Wn, (const float*)d_in[G2C1_BN], h12, h22);

    // GRU stack 3 (blocks=(b,p,g), nodes=5, adj3[g]); h = h13.
    Src s_parg = {h22 + Dc, Dc, 4, 5 * Dc, Dc, 0};
    Src s_x3   = {xs + 6 * DINc, DINc, 4, Jc * DINc, 5 * DINc, DINc};
    Src s_h13g = {h13, Dc, 1, 5 * Dc, 0, Dc};
    gru_cell(c, s_parg, s_x3, s_h13g, adj3, NGc, Bc * Pc * NGc, 5, 160,
             g3c0Wr, (const float*)d_in[G3C0_BR], g3c0Wz, (const float*)d_in[G3C0_BZ],
             g3c0Wn, (const float*)d_in[G3C0_BN], h13, cur);
    Src s_cur3 = {cur, Dc, 1, 5 * Dc, 0, Dc};
    gru_cell(c, s_cur3, ZS, s_h13g, adj3, NGc, Bc * Pc * NGc, 5, 128,
             g3c1Wr, (const float*)d_in[G3C1_BR], g3c1Wz, (const float*)d_in[G3C1_BZ],
             g3c1Wn, (const float*)d_in[G3C1_BN], h13, h23);

    { int tot = Bc * Pc * Jc * Dc;
      k_scatter_h<<<cdiv(tot, 256), 256, 0, stream>>>(H, h21, h22, h23); }
    { int tot = Bc * Pc * Jc * 3;
      k_proj_out<<<cdiv(tot, 256), 256, 0, stream>>>(H, Wout, bout, out, t); }
  }
}
